// MLDecoder_17514876633701
// MI455X (gfx1250) — matrix operand order repaired and checked by reading, not compiled, not run on hardware
//
#include <hip/hip_runtime.h>
#include <hip/hip_bf16.h>

// ---------------------------------------------------------------------------
// ML-Decoder forward for MI455X (gfx1250, wave32, WMMA).
// All heavy GEMMs run through v_wmma_f32_16x16x32_bf16 (bf16 in, f32 acc).
// Self-attention path is batch-invariant (tgt starts as broadcast query) and
// is computed once; only cross-attention K/V and later stages are per-batch.
// GEMM inner loop is branch-free: M/N tails are handled by clamping row/col
// indices (junk lands only in C entries the epilogue never writes), and K is
// always a multiple of 32 (attention K padded to 224 with zeroed pads).
// The K loop is ping-pong double-buffered (unroll x2, two named fragment
// sets) so no register copies are needed between iterations.
// ---------------------------------------------------------------------------

typedef __attribute__((ext_vector_type(16))) __bf16 v16bf;
typedef __attribute__((ext_vector_type(8)))  __bf16 v8bf;
typedef __attribute__((ext_vector_type(8)))  float  v8f;

#define NHh   8
#define Ecnt  256
#define Dm    768
#define Ff    2048
#define CINc  2048
#define HWc   196
#define Bc    64
#define HDc   96
#define DUPc  50
#define NCc   12547
#define LPAD  224            // padded att row length (multiple of 32, 16B-aligned rows)

__device__ __forceinline__ v16bf cat8(v8bf lo, v8bf hi) {
    return __builtin_shufflevector(lo, hi, 0,1,2,3,4,5,6,7,8,9,10,11,12,13,14,15);
}

// A fragment (16x32 bf16): lanes 0-15 hold K {0..7,16..23}, lanes 16-31 hold
// K {8..15,24..31} for row m = lane&15 (CDNA5 ISA 7.12.2). `row` already
// includes the per-lane koff (0 or 8 elements).
__device__ __forceinline__ v16bf load_frag_a(const __bf16* __restrict__ row, int kk) {
    v8bf lo = *(const v8bf*)(row + kk);
    v8bf hi = *(const v8bf*)(row + kk + 16);
    return cat8(lo, hi);
}

// B fragment (32x16 bf16): column n = lane&15; lanes 0-15 hold K {0..7,16..23},
// lanes 16-31 hold K {8..15,24..31}. `row` already includes the per-lane kbof (0/8).
__device__ __forceinline__ v16bf load_frag_b(const __bf16* __restrict__ row, int kk) {
    v8bf lo = *(const v8bf*)(row + kk);
    v8bf hi = *(const v8bf*)(row + kk + 16);
    return cat8(lo, hi);
}

struct GemmParams {
    const __bf16* A;  long long a_zh, a_zb; int lda;
    const __bf16* W;  long long b_zh, b_zb; int ldb;   // W = B^T, row-major [N][K]
    float*  Cf;                                        // optional f32 out
    __bf16* Cb;                                        // optional bf16 out
    __bf16* CT;                                        // optional transposed bf16 out CT[n][m]
    long long c_zh, c_zb; int ldc, ldct;
    const float* bias;                                 // optional, indexed by n
    const float* resid; int res_mod; int ld_res;       // optional, row = m % res_mod
    int M, N, K;                                       // K must be a multiple of 32
    int heads;                                         // z = h + heads*b
    float alpha;
    int relu;
};

// Block: 256 threads = 8 waves laid out 2(M) x 4(N). Block tile 128x64.
// Each wave: 4 M-subtiles x 1 N-tile of 16x16 WMMA, K stepped by 32.
__global__ __launch_bounds__(256)
void wmma_gemm_kernel(GemmParams p) {
    int z = blockIdx.z;
    int h = z % p.heads;
    int b = z / p.heads;
    const __bf16* A = p.A + (long long)h * p.a_zh + (long long)b * p.a_zb;
    const __bf16* W = p.W + (long long)h * p.b_zh + (long long)b * p.b_zb;
    long long coff  = (long long)h * p.c_zh + (long long)b * p.c_zb;

    int lane = threadIdx.x & 31;
    int wid  = threadIdx.x >> 5;
    int wm   = wid & 1;
    int wn   = wid >> 1;
    int m_wave = blockIdx.y * 128 + wm * 64;
    int n0     = blockIdx.x * 64  + wn * 16;

    int lr   = lane & 15;
    int koff = (lane >> 4) << 3;          // A lane K sub-offset: 0 or 8
    int kbof = (lane >> 4) << 3;          // B lane K sub-offset: 0 or 8

    int  n   = n0 + lr;
    bool nok = (n < p.N);
    int  ncl = nok ? n : (p.N - 1);       // clamped: load junk, never stored
    const __bf16* wrow = W + (long long)ncl * p.ldb + kbof;

    const __bf16* arow[4];
#pragma unroll
    for (int i = 0; i < 4; ++i) {
        int m   = m_wave + i * 16 + lr;
        int mcl = (m < p.M) ? m : (p.M - 1);
        arow[i] = A + (long long)mcl * p.lda + koff;
    }

    v8f acc[4] = {};
    v16bf bf0, af0[4], bf1, af1[4];

    auto loadS0 = [&](int kk) {
        bf0 = load_frag_b(wrow, kk);
#pragma unroll
        for (int i = 0; i < 4; ++i) af0[i] = load_frag_a(arow[i], kk);
    };
    auto loadS1 = [&](int kk) {
        bf1 = load_frag_b(wrow, kk);
#pragma unroll
        for (int i = 0; i < 4; ++i) af1[i] = load_frag_a(arow[i], kk);
    };
    auto mm0 = [&]() {
#pragma unroll
        for (int i = 0; i < 4; ++i)
            acc[i] = __builtin_amdgcn_wmma_f32_16x16x32_bf16(
                false, af0[i], false, bf0, (short)0, acc[i], false, false);
    };
    auto mm1 = [&]() {
#pragma unroll
        for (int i = 0; i < 4; ++i)
            acc[i] = __builtin_amdgcn_wmma_f32_16x16x32_bf16(
                false, af1[i], false, bf1, (short)0, acc[i], false, false);
    };

    int steps = p.K >> 5;                 // >= 3 in this network
    loadS0(0);
    int kk = 32;
    for (int it = 0; it < (steps - 1) >> 1; ++it) {
        loadS1(kk); kk += 32;
        mm0();
        loadS0(kk); kk += 32;
        mm1();
    }
    if ((steps & 1) == 0) {               // one more pair: S0 computed, S1 to load
        loadS1(kk);
        mm0();
        mm1();
    } else {                              // single pending step in S0
        mm0();
    }

    float bv = (p.bias && nok) ? p.bias[n] : 0.f;
    int moff = (lane >> 4) << 3;          // C: lanes 16-31 hold rows +8
#pragma unroll
    for (int i = 0; i < 4; ++i) {
        int mbase = m_wave + i * 16 + moff;
        // sub-rows r<8 from an 8-aligned base cannot cross a res_mod boundary
        long long rbase = p.resid ? (long long)(mbase % p.res_mod) * p.ld_res : 0;
#pragma unroll
        for (int r = 0; r < 8; ++r) {
            int m = mbase + r;
            if (nok && m < p.M) {
                float v = acc[i][r] * p.alpha + bv;
                if (p.relu) v = fmaxf(v, 0.f);
                if (p.resid) v += p.resid[rbase + (long long)r * p.ld_res + n];
                if (p.Cf) p.Cf[coff + (long long)m * p.ldc + n] = v;
                if (p.Cb) p.Cb[coff + (long long)m * p.ldc + n] = (__bf16)v;
                if (p.CT) p.CT[coff + (long long)n * p.ldct + m] = (__bf16)v;
            }
        }
    }
}

// ---------------------------------------------------------------------------
__global__ void f2bf_kernel(const float* __restrict__ in, __bf16* __restrict__ out,
                            long long n) {
    long long i = (long long)blockIdx.x * blockDim.x + threadIdx.x;
    long long s = (long long)gridDim.x * blockDim.x;
    for (; i < n; i += s) out[i] = (__bf16)in[i];
}

// x[b][c][p] (f32) -> xT[b*196+p][c] (bf16), LDS-tiled transpose.
__global__ __launch_bounds__(256)
void transpose_x_kernel(const float* __restrict__ x, __bf16* __restrict__ xT) {
    __shared__ float tile[32][33];
    int b = blockIdx.z;
    int c0 = blockIdx.y * 32;
    int p0 = blockIdx.x * 32;
#pragma unroll
    for (int j = 0; j < 4; ++j) {
        int c = c0 + threadIdx.y + j * 8;
        int pl = p0 + threadIdx.x;
        float v = 0.f;
        if (pl < HWc) v = x[((long long)b * CINc + c) * HWc + pl];
        tile[threadIdx.y + j * 8][threadIdx.x] = v;
    }
    __syncthreads();
#pragma unroll
    for (int j = 0; j < 4; ++j) {
        int pl = p0 + threadIdx.y + j * 8;
        int c  = c0 + threadIdx.x;
        if (pl < HWc)
            xT[((long long)b * HWc + pl) * CINc + c] =
                (__bf16)tile[threadIdx.x][threadIdx.y + j * 8];
    }
}

// Row LayerNorm over width=768; optional f32 and bf16 outputs.
__global__ __launch_bounds__(256)
void ln_kernel(const float* __restrict__ in, const float* __restrict__ g,
               const float* __restrict__ bta, float* __restrict__ outf,
               __bf16* __restrict__ outb, int width, float in_scale) {
    __shared__ float red[256];
    long long row = blockIdx.x;
    const float* xr = in + row * width;
    float local[4];
    float s = 0.f;
    int j = 0;
    for (int i = threadIdx.x; i < width; i += 256, ++j) {
        local[j] = xr[i] * in_scale;
        s += local[j];
    }
    red[threadIdx.x] = s; __syncthreads();
    for (int st = 128; st > 0; st >>= 1) {
        if (threadIdx.x < st) red[threadIdx.x] += red[threadIdx.x + st];
        __syncthreads();
    }
    float mean = red[0] / width; __syncthreads();
    float vs = 0.f;
    j = 0;
    for (int i = threadIdx.x; i < width; i += 256, ++j) {
        float d = local[j] - mean; vs += d * d;
    }
    red[threadIdx.x] = vs; __syncthreads();
    for (int st = 128; st > 0; st >>= 1) {
        if (threadIdx.x < st) red[threadIdx.x] += red[threadIdx.x + st];
        __syncthreads();
    }
    float rstd = rsqrtf(red[0] / width + 1e-5f);
    j = 0;
    for (int i = threadIdx.x; i < width; i += 256, ++j) {
        float y = g[i] * (local[j] - mean) * rstd + bta[i];
        if (outf) outf[row * width + i] = y;
        if (outb) outb[row * width + i] = (__bf16)y;
    }
}

// In-place row softmax on bf16 att rows (L <= 256), pad cols [L,ld) zeroed.
__global__ __launch_bounds__(256)
void softmax_kernel(__bf16* __restrict__ att, int L, int ld) {
    __shared__ float red[256];
    long long row = blockIdx.x;
    __bf16* r = att + row * ld;
    int tid = threadIdx.x;
    float v = (tid < L) ? (float)r[tid] : -3.4e38f;
    red[tid] = v; __syncthreads();
    for (int st = 128; st > 0; st >>= 1) {
        if (tid < st) red[tid] = fmaxf(red[tid], red[tid + st]);
        __syncthreads();
    }
    float mx = red[0]; __syncthreads();
    float e = (tid < L) ? __expf(v - mx) : 0.f;
    red[tid] = e; __syncthreads();
    for (int st = 128; st > 0; st >>= 1) {
        if (tid < st) red[tid] += red[tid + st];
        __syncthreads();
    }
    float inv = 1.f / red[0];
    if (tid < ld) r[tid] = (__bf16)((tid < L) ? e * inv : 0.f);
}

// GroupFC: block = one (b,e); logits[b][e*50+f] = h[b,e,:].dp[e,:,f] + bias.
__global__ __launch_bounds__(64)
void groupfc_kernel(const __bf16* __restrict__ h, const float* __restrict__ dp,
                    const float* __restrict__ bias, float* __restrict__ out) {
    __shared__ float hrow[Dm];
    int be = blockIdx.x;
    int e  = be & (Ecnt - 1);
    const __bf16* hr = h + (long long)be * Dm;
    for (int i = threadIdx.x; i < Dm; i += 64) hrow[i] = (float)hr[i];
    __syncthreads();
    int f = threadIdx.x;
    if (f < DUPc) {
        const float* dpe = dp + ((long long)e * Dm) * DUPc + f;
        float s = 0.f;
        for (int d = 0; d < Dm; ++d) s += hrow[d] * dpe[(long long)d * DUPc];
        int cls = e * DUPc + f;
        if (cls < NCc) {
            int b = be >> 8;
            out[(long long)b * NCc + cls] = s + bias[cls];
        }
    }
}

// ---------------------------------------------------------------------------
extern "C" void kernel_launch(void* const* d_in, const int* in_sizes, int n_in,
                              void* d_out, int out_size, void* d_ws, size_t ws_size,
                              hipStream_t stream) {
    (void)in_sizes; (void)n_in; (void)out_size; (void)ws_size;
    const float* x       = (const float*)d_in[0];
    const float* embed_w = (const float*)d_in[1];
    const float* embed_b = (const float*)d_in[2];
    const float* query   = (const float*)d_in[3];
    const float* sa_wq = (const float*)d_in[4],  *sa_wk = (const float*)d_in[5];
    const float* sa_wv = (const float*)d_in[6];
    const float* sa_bq = (const float*)d_in[7],  *sa_bk = (const float*)d_in[8];
    const float* sa_bv = (const float*)d_in[9];
    const float* sa_wo = (const float*)d_in[10], *sa_bo = (const float*)d_in[11];
    const float* ca_wq = (const float*)d_in[12], *ca_wk = (const float*)d_in[13];
    const float* ca_wv = (const float*)d_in[14];
    const float* ca_bq = (const float*)d_in[15], *ca_bk = (const float*)d_in[16];
    const float* ca_bv = (const float*)d_in[17];
    const float* ca_wo = (const float*)d_in[18], *ca_bo = (const float*)d_in[19];
    const float* lin1_w = (const float*)d_in[20], *lin1_b = (const float*)d_in[21];
    const float* lin2_w = (const float*)d_in[22], *lin2_b = (const float*)d_in[23];
    const float* ln1_g = (const float*)d_in[24], *ln1_b = (const float*)d_in[25];
    const float* ln2_g = (const float*)d_in[26], *ln2_b = (const float*)d_in[27];
    const float* ln3_g = (const float*)d_in[28], *ln3_b = (const float*)d_in[29];
    const float* ln4_g = (const float*)d_in[30], *ln4_b = (const float*)d_in[31];
    const float* dp      = (const float*)d_in[32];
    const float* dp_bias = (const float*)d_in[33];
    float* out = (float*)d_out;

    char* ws = (char*)d_ws;
    size_t off = 0;
    auto alloc = [&](size_t bytes) -> void* {
        void* p = ws + off;
        off = (off + bytes + 255) & ~(size_t)255;
        return p;
    };
    const long long TOK = (long long)Bc * HWc;      // 12544
    const long long BE  = (long long)Bc * Ecnt;     // 16384

    __bf16* wb_embed = (__bf16*)alloc((size_t)Dm * CINc * 2);
    __bf16* wb_saq = (__bf16*)alloc((size_t)Dm * Dm * 2);
    __bf16* wb_sak = (__bf16*)alloc((size_t)Dm * Dm * 2);
    __bf16* wb_sav = (__bf16*)alloc((size_t)Dm * Dm * 2);
    __bf16* wb_sao = (__bf16*)alloc((size_t)Dm * Dm * 2);
    __bf16* wb_caq = (__bf16*)alloc((size_t)Dm * Dm * 2);
    __bf16* wb_cak = (__bf16*)alloc((size_t)Dm * Dm * 2);
    __bf16* wb_cav = (__bf16*)alloc((size_t)Dm * Dm * 2);
    __bf16* wb_cao = (__bf16*)alloc((size_t)Dm * Dm * 2);
    __bf16* wb_l1  = (__bf16*)alloc((size_t)Ff * Dm * 2);
    __bf16* wb_l2  = (__bf16*)alloc((size_t)Dm * Ff * 2);
    __bf16* xT     = (__bf16*)alloc((size_t)TOK * CINc * 2);
    __bf16* emb    = (__bf16*)alloc((size_t)TOK * Dm * 2);
    float*  tgt1f  = (float*) alloc((size_t)Ecnt * Dm * 4);
    __bf16* tgt1b  = (__bf16*)alloc((size_t)Ecnt * Dm * 2);
    __bf16* saq    = (__bf16*)alloc((size_t)Ecnt * Dm * 2);
    __bf16* sak    = (__bf16*)alloc((size_t)Ecnt * Dm * 2);
    __bf16* savT   = (__bf16*)alloc((size_t)Dm * Ecnt * 2);
    __bf16* saatt  = (__bf16*)alloc((size_t)NHh * Ecnt * Ecnt * 2);
    __bf16* sao    = (__bf16*)alloc((size_t)Ecnt * Dm * 2);
    float*  sapre  = (float*) alloc((size_t)Ecnt * Dm * 4);
    float*  tgt2f  = (float*) alloc((size_t)Ecnt * Dm * 4);
    __bf16* tgt2b  = (__bf16*)alloc((size_t)Ecnt * Dm * 2);
    __bf16* caq    = (__bf16*)alloc((size_t)Ecnt * Dm * 2);
    __bf16* cak    = (__bf16*)alloc((size_t)TOK * Dm * 2);
    __bf16* cavT   = (__bf16*)alloc((size_t)Dm * Bc * LPAD * 2);
    __bf16* caatt  = (__bf16*)alloc((size_t)Bc * NHh * Ecnt * LPAD * 2);
    __bf16* cao    = (__bf16*)alloc((size_t)BE * Dm * 2);
    float*  capre  = (float*) alloc((size_t)BE * Dm * 4);
    float*  tgt3f  = (float*) alloc((size_t)BE * Dm * 4);
    __bf16* tgt3b  = (__bf16*)alloc((size_t)BE * Dm * 2);
    __bf16* h1     = (__bf16*)alloc((size_t)BE * Ff * 2);
    float*  ffpre  = (float*) alloc((size_t)BE * Dm * 4);
    __bf16* hfin   = (__bf16*)alloc((size_t)BE * Dm * 2);

    auto f2bf = [&](const float* in, __bf16* o, long long n) {
        int grid = (int)((n + 255) / 256); if (grid > 4096) grid = 4096;
        f2bf_kernel<<<grid, 256, 0, stream>>>(in, o, n);
    };
    auto gemm = [&](const __bf16* A, long long azh, long long azb, int lda,
                    const __bf16* W, long long bzh, long long bzb, int ldb,
                    float* Cf, __bf16* Cb, __bf16* CT,
                    long long czh, long long czb, int ldc, int ldct,
                    const float* bias, const float* resid, int res_mod, int ld_res,
                    int M, int N, int K, int heads, int Z, float alpha, int relu) {
        GemmParams p{A, azh, azb, lda, W, bzh, bzb, ldb, Cf, Cb, CT,
                     czh, czb, ldc, ldct, bias, resid, res_mod, ld_res,
                     M, N, K, heads, alpha, relu};
        dim3 grid((N + 63) / 64, (M + 127) / 128, Z);
        wmma_gemm_kernel<<<grid, 256, 0, stream>>>(p);
    };
    const float sc = 0.102062072616f;   // 1/sqrt(96)

    // Weight conversion
    f2bf(embed_w, wb_embed, (long long)Dm * CINc);
    f2bf(sa_wq, wb_saq, (long long)Dm * Dm);  f2bf(sa_wk, wb_sak, (long long)Dm * Dm);
    f2bf(sa_wv, wb_sav, (long long)Dm * Dm);  f2bf(sa_wo, wb_sao, (long long)Dm * Dm);
    f2bf(ca_wq, wb_caq, (long long)Dm * Dm);  f2bf(ca_wk, wb_cak, (long long)Dm * Dm);
    f2bf(ca_wv, wb_cav, (long long)Dm * Dm);  f2bf(ca_wo, wb_cao, (long long)Dm * Dm);
    f2bf(lin1_w, wb_l1, (long long)Ff * Dm);  f2bf(lin2_w, wb_l2, (long long)Dm * Ff);

    // Zero v^T pad columns (K range 196..223 must contribute 0 in att@v GEMM).
    hipMemsetAsync(cavT, 0, (size_t)Dm * Bc * LPAD * 2, stream);

    // x transpose + embed GEMM (+bias+ReLU)
    transpose_x_kernel<<<dim3(7, CINc / 32, Bc), dim3(32, 8), 0, stream>>>(x, xT);
    gemm(xT, 0, 0, CINc, wb_embed, 0, 0, CINc, nullptr, emb, nullptr, 0, 0, Dm, 0,
         embed_b, nullptr, 1, Dm, (int)TOK, Dm, CINc, 1, 1, 1.f, 1);

    // tgt1 = LN(2*query)   [batch-invariant]
    ln_kernel<<<Ecnt, 256, 0, stream>>>(query, ln1_g, ln1_b, tgt1f, tgt1b, Dm, 2.f);

    // Self-attention (computed once for all batches)
    gemm(tgt1b, 0, 0, Dm, wb_saq, 0, 0, Dm, nullptr, saq, nullptr, 0, 0, Dm, 0,
         sa_bq, nullptr, 1, Dm, Ecnt, Dm, Dm, 1, 1, 1.f, 0);
    gemm(tgt1b, 0, 0, Dm, wb_sak, 0, 0, Dm, nullptr, sak, nullptr, 0, 0, Dm, 0,
         sa_bk, nullptr, 1, Dm, Ecnt, Dm, Dm, 1, 1, 1.f, 0);
    gemm(tgt1b, 0, 0, Dm, wb_sav, 0, 0, Dm, nullptr, nullptr, savT, 0, 0, 0, Ecnt,
         sa_bv, nullptr, 1, Dm, Ecnt, Dm, Dm, 1, 1, 1.f, 0);
    gemm(saq, HDc, 0, Dm, sak, HDc, 0, Dm, nullptr, saatt, nullptr,
         (long long)Ecnt * Ecnt, 0, Ecnt, 0, nullptr, nullptr, 1, Dm,
         Ecnt, Ecnt, HDc, NHh, NHh, sc, 0);
    softmax_kernel<<<NHh * Ecnt, 256, 0, stream>>>(saatt, Ecnt, Ecnt);
    gemm(saatt, (long long)Ecnt * Ecnt, 0, Ecnt, savT, (long long)HDc * Ecnt, 0, Ecnt,
         nullptr, sao, nullptr, HDc, 0, Dm, 0, nullptr, nullptr, 1, Dm,
         Ecnt, HDc, Ecnt, NHh, NHh, 1.f, 0);
    gemm(sao, 0, 0, Dm, wb_sao, 0, 0, Dm, sapre, nullptr, nullptr, 0, 0, Dm, 0,
         sa_bo, tgt1f, Ecnt, Dm, Ecnt, Dm, Dm, 1, 1, 1.f, 0);
    ln_kernel<<<Ecnt, 256, 0, stream>>>(sapre, ln2_g, ln2_b, tgt2f, tgt2b, Dm, 1.f);

    // Cross-attention
    gemm(tgt2b, 0, 0, Dm, wb_caq, 0, 0, Dm, nullptr, caq, nullptr, 0, 0, Dm, 0,
         ca_bq, nullptr, 1, Dm, Ecnt, Dm, Dm, 1, 1, 1.f, 0);
    gemm(emb, 0, 0, Dm, wb_cak, 0, 0, Dm, nullptr, cak, nullptr, 0, 0, Dm, 0,
         ca_bk, nullptr, 1, Dm, (int)TOK, Dm, Dm, 1, 1, 1.f, 0);
    gemm(emb, 0, (long long)HWc * Dm, Dm, wb_cav, 0, 0, Dm, nullptr, nullptr, cavT,
         0, LPAD, 0, Bc * LPAD, ca_bv, nullptr, 1, Dm, HWc, Dm, Dm, 1, Bc, 1.f, 0);
    gemm(caq, HDc, 0, Dm, cak, HDc, (long long)HWc * Dm, Dm, nullptr, caatt, nullptr,
         (long long)Ecnt * LPAD, (long long)NHh * Ecnt * LPAD, LPAD, 0,
         nullptr, nullptr, 1, Dm, Ecnt, HWc, HDc, NHh, Bc * NHh, sc, 0);
    softmax_kernel<<<Bc * NHh * Ecnt, 256, 0, stream>>>(caatt, HWc, LPAD);
    // att@v with K padded to 224: att pad cols are zero, cavT pads zeroed.
    gemm(caatt, (long long)Ecnt * LPAD, (long long)NHh * Ecnt * LPAD, LPAD,
         cavT, (long long)HDc * Bc * LPAD, LPAD, Bc * LPAD,
         nullptr, cao, nullptr, HDc, (long long)Ecnt * Dm, Dm, 0,
         nullptr, nullptr, 1, Dm, Ecnt, HDc, LPAD, NHh, Bc * NHh, 1.f, 0);
    gemm(cao, 0, 0, Dm, wb_cao, 0, 0, Dm, capre, nullptr, nullptr, 0, 0, Dm, 0,
         ca_bo, tgt2f, Ecnt, Dm, (int)BE, Dm, Dm, 1, 1, 1.f, 0);
    ln_kernel<<<(int)BE, 256, 0, stream>>>(capre, ln3_g, ln3_b, tgt3f, tgt3b, Dm, 1.f);

    // FFN
    gemm(tgt3b, 0, 0, Dm, wb_l1, 0, 0, Dm, nullptr, h1, nullptr, 0, 0, Ff, 0,
         lin1_b, nullptr, 1, Dm, (int)BE, Ff, Dm, 1, 1, 1.f, 1);
    gemm(h1, 0, 0, Ff, wb_l2, 0, 0, Ff, ffpre, nullptr, nullptr, 0, 0, Dm, 0,
         lin2_b, tgt3f, (int)BE, Dm, (int)BE, Dm, Ff, 1, 1, 1.f, 0);
    ln_kernel<<<(int)BE, 256, 0, stream>>>(ffpre, ln4_g, ln4_b, nullptr, hfin, Dm, 1.f);

    // GroupFC -> logits
    groupfc_kernel<<<(int)BE, 64, 0, stream>>>(hfin, dp, dp_bias, out);
}
